// HyperbolicAttentionLayer_30124900614682
// MI455X (gfx1250) — matrix operand order repaired and checked by reading, not compiled, not run on hardware
//
#include <hip/hip_runtime.h>
#include <hip/hip_bf16.h>
#include <math.h>

#define N_NODES 20000
#define DIM     512
#define N_EDGES 320000
#define EPSV    1e-7f

typedef __attribute__((ext_vector_type(16))) __bf16 v16bf;
typedef __attribute__((ext_vector_type(8)))  __bf16 v8bf;
typedef __attribute__((ext_vector_type(8)))  float  v8f;
typedef __attribute__((ext_vector_type(4)))  int    v4i;

// pointer-to-AS typedefs for the async global->LDS builtin
typedef __attribute__((address_space(1))) v4i* gv4i_p;  // global int4*
typedef __attribute__((address_space(3))) v4i* sv4i_p;  // LDS int4*

#if defined(__has_builtin)
#if __has_builtin(__builtin_amdgcn_global_load_async_to_lds_b128) && \
    __has_builtin(__builtin_amdgcn_s_wait_asynccnt)
#define HAVE_ASYNC_LDS 1
#endif
#endif
#ifndef HAVE_ASYNC_LDS
#define HAVE_ASYNC_LDS 0
#endif

// ---- monotone int encoding of float for atomicMax-based segment max ----
__device__ __forceinline__ int enc_f32(float f) {
    int b = __float_as_int(f);
    return (b >= 0) ? b : (b ^ 0x7FFFFFFF);
}
__device__ __forceinline__ float dec_f32(int k) {
    int b = (k >= 0) ? k : (k ^ 0x7FFFFFFF);
    return __int_as_float(b);
}

// ---------------- init: attn=0, denom=0, m_enc=INT_MIN ----------------
__global__ void init_kernel(float* __restrict__ attn, float* __restrict__ denom,
                            int* __restrict__ m_enc) {
    int idx = blockIdx.x * blockDim.x + threadIdx.x;
    if (idx < N_NODES * DIM) attn[idx] = 0.0f;
    if (idx < N_NODES) {
        denom[idx] = 0.0f;
        m_enc[idx] = (int)0x80000000;  // INT_MIN: lowest key, below enc(-inf)
    }
}

// ---------------- logmap0: t = artanh(||x||)/||x|| * x, store bf16 ----------
__global__ void logmap_kernel(const float* __restrict__ x, __bf16* __restrict__ tb) {
    __shared__ float red[256];
    const int row = blockIdx.x;
    const int tid = threadIdx.x;
    const float* xr = x + (size_t)row * DIM;
    float a = xr[tid];
    float b = xr[tid + 256];
    red[tid] = a * a + b * b;
    __syncthreads();
    for (int ofs = 128; ofs > 0; ofs >>= 1) {
        if (tid < ofs) red[tid] += red[tid + ofs];
        __syncthreads();
    }
    float n  = sqrtf(red[0]);
    float nc = fminf(fmaxf(n, EPSV), 1.0f - 1e-6f);
    float at = 0.5f * logf((1.0f + nc) / (1.0f - nc));   // artanh(nc)
    float sc = at / nc;
    tb[(size_t)row * DIM + tid]       = (__bf16)(a * sc);
    tb[(size_t)row * DIM + tid + 256] = (__bf16)(b * sc);
}

// ---- weights fp32 -> bf16 (Wq*qscale|Wk|Wv|Wo packed); fold q-scale in -----
__global__ void wconv_kernel(const float* __restrict__ Wq, const float* __restrict__ Wk,
                             const float* __restrict__ Wv, const float* __restrict__ Wo,
                             __bf16* __restrict__ wb, float qscale) {
    int idx = blockIdx.x * blockDim.x + threadIdx.x;
    if (idx >= DIM * DIM) return;
    wb[idx]                 = (__bf16)(Wq[idx] * qscale);
    wb[DIM * DIM + idx]     = (__bf16)Wk[idx];
    wb[2 * DIM * DIM + idx] = (__bf16)Wv[idx];
    wb[3 * DIM * DIM + idx] = (__bf16)Wo[idx];
}

// ---- biases packed [bq*qscale | bk | bv | bo] ------------------------------
__global__ void biaspack_kernel(const float* __restrict__ bq, const float* __restrict__ bk,
                                const float* __restrict__ bv, const float* __restrict__ bo,
                                float* __restrict__ bp, float qscale) {
    int i = blockIdx.x * blockDim.x + threadIdx.x;
    if (i >= DIM) return;
    bp[i]           = bq[i] * qscale;
    bp[DIM + i]     = bk[i];
    bp[2 * DIM + i] = bv[i];
    bp[3 * DIM + i] = bo[i];
}

// ---------------- fused WMMA GEMM: for w in [0,NW): Cw = A @ Ww^T + bw -------
// Block = 256 threads = 8 waves. Wave holds rows [m0,m0+16); block cols [j0,j0+64).
// B-tiles (NW x 64 x 512 bf16) staged once per block into LDS via async loads;
// A fragments stream from global and feed NW*4 WMMAs per K-step.
template <int NW>
__global__ void __launch_bounds__(256) gemm_wmma_kernel(
    const __bf16* __restrict__ Ab, const __bf16* __restrict__ Wb,
    const float* __restrict__ biases,
    float* __restrict__ o0, float* __restrict__ o1, float* __restrict__ o2,
    int nRows) {
    extern __shared__ __bf16 smem[];  // [NW][64][DIM]

    const int lane = threadIdx.x & 31;
    const int wave = threadIdx.x >> 5;
    const int m0 = blockIdx.x * 128 + wave * 16;
    const int j0 = blockIdx.y * 64;

    // ---- stage B tiles: rows [j0, j0+64) of each W are contiguous slabs ----
#pragma unroll
    for (int w = 0; w < NW; ++w) {
        const __bf16* srcW = Wb + (size_t)w * DIM * DIM + (size_t)j0 * DIM;
        __bf16* dstL = smem + (size_t)w * 64 * DIM;
        for (int i = threadIdx.x; i < 64 * DIM / 8; i += 256) {
#if HAVE_ASYNC_LDS
            __builtin_amdgcn_global_load_async_to_lds_b128(
                (gv4i_p)(srcW + i * 8), (sv4i_p)(dstL + i * 8), 0, 0);
#else
            *(v8bf*)(dstL + i * 8) = *(const v8bf*)(srcW + i * 8);
#endif
        }
    }
#if HAVE_ASYNC_LDS
    __builtin_amdgcn_s_wait_asynccnt(0);
#endif
    __syncthreads();

    const int rl = lane & 15;
    int rA = m0 + rl;
    if (rA > nRows - 1) rA = nRows - 1;           // clamp: EXEC must stay full
    const int ksel   = (lane < 16) ? 0 : 16;      // A fragment K-half select
    const int kbaseB = (lane < 16) ? 0 : 16;      // B fragment K-half select

    const __bf16* arow = Ab + (size_t)rA * DIM;

    v8f acc[NW][4];
#pragma unroll
    for (int w = 0; w < NW; ++w)
#pragma unroll
        for (int t = 0; t < 4; ++t) acc[w][t] = (v8f){};

    for (int k0 = 0; k0 < DIM; k0 += 32) {
        // A fragment: lanes 0-15 hold K=k0..k0+15, lanes 16-31 hold K=k0+16..k0+31
        v16bf a = *(const v16bf*)(arow + k0 + ksel);

#pragma unroll
        for (int w = 0; w < NW; ++w) {
#pragma unroll
            for (int t = 0; t < 4; ++t) {
                // B: lanes 0-15 hold K=k0..k0+15, lanes 16-31 hold K=k0+16..k0+31
                const __bf16* bp =
                    smem + ((size_t)(w * 64 + t * 16 + rl) * DIM) + kbaseB + k0;
                v16bf b = *(const v16bf*)bp;
                acc[w][t] = __builtin_amdgcn_wmma_f32_16x16x32_bf16(
                    false, a, false, b, (short)0, acc[w][t], false, false);
            }
        }
    }

    // C/D layout: VGPR r -> M = r (lanes 0-15) or r+8 (lanes 16-31), N = lane&15
    float* outs[3] = {o0, o1, o2};
    const int rowbase = m0 + ((lane >= 16) ? 8 : 0);
#pragma unroll
    for (int w = 0; w < NW; ++w) {
        const float bs0 = biases[w * DIM + j0 + rl];
        const float bs1 = biases[w * DIM + j0 + 16 + rl];
        const float bs2 = biases[w * DIM + j0 + 32 + rl];
        const float bs3 = biases[w * DIM + j0 + 48 + rl];
        float* out = outs[w];
#pragma unroll
        for (int r = 0; r < 8; r++) {
            int row = rowbase + r;
            if (row < nRows) {
                float* crow = out + (size_t)row * DIM + j0 + rl;
                crow[0]  = acc[w][0][r] + bs0;
                crow[16] = acc[w][1][r] + bs1;
                crow[32] = acc[w][2][r] + bs2;
                crow[48] = acc[w][3][r] + bs3;
            }
        }
    }
}

// ---------------- edge scores: score[e] = dot(q[dst], k[src]); segment max ---
__global__ void __launch_bounds__(256) edge_score_kernel(
    const float* __restrict__ q, const float* __restrict__ k,
    const int* __restrict__ src, const int* __restrict__ dst,
    float* __restrict__ escore, int* __restrict__ m_enc) {
    int e = blockIdx.x * 8 + (threadIdx.x >> 5);
    if (e >= N_EDGES) return;
    int lane = threadIdx.x & 31;
    const float4* qp = (const float4*)(q + (size_t)dst[e] * DIM);
    const float4* kp = (const float4*)(k + (size_t)src[e] * DIM);
    float s = 0.0f;
#pragma unroll
    for (int i = 0; i < 4; i++) {
        float4 a = qp[lane + 32 * i];
        float4 b = kp[lane + 32 * i];
        s += a.x * b.x + a.y * b.y + a.z * b.z + a.w * b.w;
    }
#pragma unroll
    for (int off = 16; off > 0; off >>= 1) s += __shfl_xor(s, off, 32);
    if (lane == 0) {
        escore[e] = s;
        atomicMax(m_enc + dst[e], enc_f32(s));
    }
}

// ---------------- e = exp(score - m[dst]); denom[dst] += e ------------------
__global__ void edge_exp_kernel(const int* __restrict__ dst, float* __restrict__ escore,
                                const int* __restrict__ m_enc, float* __restrict__ denom) {
    int e = blockIdx.x * blockDim.x + threadIdx.x;
    if (e >= N_EDGES) return;
    int d = dst[e];
    float ex = expf(escore[e] - dec_f32(m_enc[d]));
    escore[e] = ex;
    atomicAdd(denom + d, ex);
}

// ---------------- attn[dst] += alpha * v[src] -------------------------------
__global__ void __launch_bounds__(256) edge_scatter_kernel(
    const float* __restrict__ v, const int* __restrict__ src, const int* __restrict__ dst,
    const float* __restrict__ escore, const float* __restrict__ denom,
    float* __restrict__ attn) {
    int e = blockIdx.x * 8 + (threadIdx.x >> 5);
    if (e >= N_EDGES) return;
    int lane = threadIdx.x & 31;
    int d = dst[e];
    float alpha = escore[e] / denom[d];
    const float4* vp = (const float4*)(v + (size_t)src[e] * DIM);
    float* ap = attn + (size_t)d * DIM;
#pragma unroll
    for (int i = 0; i < 4; i++) {
        int c4 = lane + 32 * i;
        float4 vv = vp[c4];
        atomicAdd(ap + 4 * c4 + 0, alpha * vv.x);
        atomicAdd(ap + 4 * c4 + 1, alpha * vv.y);
        atomicAdd(ap + 4 * c4 + 2, alpha * vv.z);
        atomicAdd(ap + 4 * c4 + 3, alpha * vv.w);
    }
}

// ---------------- fp32 -> bf16 ----------------------------------------------
__global__ void f32_to_bf16_kernel(const float* __restrict__ in, __bf16* __restrict__ out, int n) {
    int i = blockIdx.x * blockDim.x + threadIdx.x;
    if (i < n) out[i] = (__bf16)in[i];
}

// ---------------- expmap0: out = tanh(||h||)/||h|| * h ----------------------
__global__ void expmap_kernel(const float* __restrict__ h, float* __restrict__ out) {
    __shared__ float red[256];
    const int row = blockIdx.x;
    const int tid = threadIdx.x;
    const float* hr = h + (size_t)row * DIM;
    float a = hr[tid];
    float b = hr[tid + 256];
    red[tid] = a * a + b * b;
    __syncthreads();
    for (int ofs = 128; ofs > 0; ofs >>= 1) {
        if (tid < ofs) red[tid] += red[tid + ofs];
        __syncthreads();
    }
    float n  = sqrtf(red[0]);
    float nc = fmaxf(n, EPSV);
    float sc = tanhf(nc) / nc;
    out[(size_t)row * DIM + tid]       = a * sc;
    out[(size_t)row * DIM + tid + 256] = b * sc;
}

extern "C" void kernel_launch(void* const* d_in, const int* in_sizes, int n_in,
                              void* d_out, int out_size, void* d_ws, size_t ws_size,
                              hipStream_t stream) {
    const float* x  = (const float*)d_in[0];
    const int* src  = (const int*)d_in[1];
    const int* dst  = (const int*)d_in[2];
    const float* Wq = (const float*)d_in[3];
    const float* bq = (const float*)d_in[4];
    const float* Wk = (const float*)d_in[5];
    const float* bk = (const float*)d_in[6];
    const float* Wv = (const float*)d_in[7];
    const float* bv = (const float*)d_in[8];
    const float* Wo = (const float*)d_in[9];
    const float* bo = (const float*)d_in[10];
    (void)in_sizes; (void)n_in; (void)out_size; (void)ws_size;

    char* ws = (char*)d_ws;
    size_t off = 0;
    auto carve = [&](size_t bytes) -> void* {
        void* p = ws + off;
        off += (bytes + 255) & ~(size_t)255;
        return p;
    };
    __bf16* tb    = (__bf16*)carve((size_t)N_NODES * DIM * 2);   // tangent bf16; reused for attn-bf16
    __bf16* wb    = (__bf16*)carve((size_t)4 * DIM * DIM * 2);   // Wq*s|Wk|Wv|Wo bf16
    float* bp     = (float*)carve((size_t)4 * DIM * 4);          // packed biases
    float* q      = (float*)carve((size_t)N_NODES * DIM * 4);    // reused as h
    float* kk     = (float*)carve((size_t)N_NODES * DIM * 4);
    float* v      = (float*)carve((size_t)N_NODES * DIM * 4);
    float* attn   = (float*)carve((size_t)N_NODES * DIM * 4);
    float* escore = (float*)carve((size_t)N_EDGES * 4);
    int* m_enc    = (int*)carve((size_t)N_NODES * 4);
    float* denom  = (float*)carve((size_t)N_NODES * 4);

    const float qscale = 1.0f / sqrtf((float)DIM);
    dim3 blk(256);
    init_kernel<<<(N_NODES * DIM + 255) / 256, blk, 0, stream>>>(attn, denom, m_enc);
    logmap_kernel<<<N_NODES, blk, 0, stream>>>(x, tb);
    wconv_kernel<<<(DIM * DIM + 255) / 256, blk, 0, stream>>>(Wq, Wk, Wv, Wo, wb, qscale);
    biaspack_kernel<<<(DIM + 255) / 256, blk, 0, stream>>>(bq, bk, bv, bo, bp, qscale);

    dim3 ggrid((N_NODES + 127) / 128, DIM / 64);
    // Fused QKV GEMM: 3 B-tiles (192 KB) staged in LDS, A streamed once.
    gemm_wmma_kernel<3><<<ggrid, blk, 3 * 64 * DIM * sizeof(__bf16), stream>>>(
        tb, wb, bp, q, kk, v, N_NODES);

    edge_score_kernel<<<(N_EDGES + 7) / 8, blk, 0, stream>>>(q, kk, src, dst, escore, m_enc);
    edge_exp_kernel<<<(N_EDGES + 255) / 256, blk, 0, stream>>>(dst, escore, m_enc, denom);
    edge_scatter_kernel<<<(N_EDGES + 7) / 8, blk, 0, stream>>>(v, src, dst, escore, denom, attn);

    f32_to_bf16_kernel<<<(N_NODES * DIM + 255) / 256, blk, 0, stream>>>(attn, tb, N_NODES * DIM);
    float* h = q;  // q no longer needed; reuse as output-projection result
    gemm_wmma_kernel<1><<<ggrid, blk, 64 * DIM * sizeof(__bf16), stream>>>(
        tb, wb + (size_t)3 * DIM * DIM, bp + 3 * DIM, h, nullptr, nullptr, N_NODES);

    expmap_kernel<<<N_NODES, blk, 0, stream>>>(h, (float*)d_out);
}
